// EigenvalueSegmentPrec_51247549776380
// MI455X (gfx1250) — matrix operand order repaired and checked by reading, not compiled, not run on hardware
//
#include <hip/hip_runtime.h>
#include <hip/hip_bf16.h>
#include <cstdint>

// ---------------------------------------------------------------------------
// Types for CDNA5 WMMA (wave32): v16bf A/B fragments, v8f accumulator.
// ---------------------------------------------------------------------------
typedef __attribute__((ext_vector_type(16))) __bf16 v16bf;
typedef __attribute__((ext_vector_type(8)))  __bf16 v8bf;
typedef __attribute__((ext_vector_type(8)))  float  v8f;

union V16U { v16bf v; v8bf h[2]; };
union BF8U { v8bf v; unsigned short u[8]; uint4 q; };

__device__ __forceinline__ unsigned short f2bf(float f) {
  union { float f; unsigned int u; } c; c.f = f;
  unsigned int u = c.u;
  return (unsigned short)((u + 0x7FFFu + ((u >> 16) & 1u)) >> 16);  // RNE
}

__device__ __forceinline__ unsigned lds_addr(const void* p) {
  // generic shared pointer: low 32 bits are the wave-relative LDS address
  return (unsigned)(uintptr_t)p;
}

// ---------------------------------------------------------------------------
// Generic bf16-WMMA GEMM:  C(M x Nc) = A(M x K) @ B(K x Nc)
//   A: row-major; fp32 (cvt in-register, K%4==0) or bf16 (K-padded to 32).
//   B: TRANSPOSED bf16 Bt (Nc x K, ld ldb, K-padded, zero tail).
//   128 threads (4 waves), 64x64 tile, wave = 32x32.
//   bf16-A: 3-deep global_load_async_to_lds_b128 pipeline (ASYNCcnt).
//   fp32-A: register-pipelined (manually 2x-unrolled, spill-free) staging.
//   Epilogue (optional): +bias[n], ReLU, out (=|+=) cw[depth*64+n]*val,
//     transposed bf16 store stT[n][m], row-major bf16 store stRM[m][n].
// ---------------------------------------------------------------------------
template<bool A_F32>
__global__ __launch_bounds__(128)
void gemm_wmma_bf16(const void* __restrict__ Ap, int lda,
                    const unsigned short* __restrict__ Bt, int ldb,
                    int M, int K,
                    const float* __restrict__ bias, int relu,
                    float* __restrict__ out, const float* __restrict__ cw,
                    int depth, int out_mode,
                    unsigned short* __restrict__ stT, int ldT,
                    unsigned short* __restrict__ stRM, int ldRM)
{
  constexpr int ALD = 40;  // padded LDS stride (halves): 80B, conflict-free
  constexpr int BLD = 40;
  __shared__ unsigned short As[3][64 * ALD];
  __shared__ unsigned short Bs[3][64 * BLD];

  const int tid   = threadIdx.x;
  const int lane  = tid & 31;
  const int wave  = tid >> 5;
  const int mtile = blockIdx.x * 64;
  const int ntile = blockIdx.y * 64;
  const int wm    = (wave & 1) * 32;
  const int wn    = (wave >> 1) * 32;

  const v8f vzero = {0.f, 0.f, 0.f, 0.f, 0.f, 0.f, 0.f, 0.f};
  v8f acc[2][2];
  #pragma unroll
  for (int i = 0; i < 2; ++i)
    #pragma unroll
    for (int j = 0; j < 2; ++j) acc[i][j] = vzero;

  const int nk = (K + 31) >> 5;
  const int l15 = lane & 15;
  const int hi  = lane >> 4;

  // ---- WMMA compute from LDS buffer b ----
  auto compute = [&](int b) {
    V16U aF[2], bF[2];
    #pragma unroll
    for (int mi = 0; mi < 2; ++mi) {
      int r = wm + mi * 16 + l15;
      int ka = hi * 8;                 // A: K = {0..7,16..23} / {8..15,24..31}
      aF[mi].h[0] = *(const v8bf*)&As[b][r * ALD + ka];
      aF[mi].h[1] = *(const v8bf*)&As[b][r * ALD + ka + 16];
    }
    #pragma unroll
    for (int ni = 0; ni < 2; ++ni) {
      int r = wn + ni * 16 + l15;
      int kb = hi * 8;                 // B: K = {0..7,16..23} / {8..15,24..31}
      bF[ni].h[0] = *(const v8bf*)&Bs[b][r * BLD + kb];
      bF[ni].h[1] = *(const v8bf*)&Bs[b][r * BLD + kb + 16];
    }
    #pragma unroll
    for (int mi = 0; mi < 2; ++mi)
      #pragma unroll
      for (int ni = 0; ni < 2; ++ni)
        acc[mi][ni] = __builtin_amdgcn_wmma_f32_16x16x32_bf16(
            false, aF[mi].v, false, bF[ni].v, (short)0, acc[mi][ni],
            false, false);
  };

  if constexpr (!A_F32) {
    // ========== bf16-A: 3-deep async DMA pipeline (4 ops/tile/wave) ==========
    const unsigned short* A = (const unsigned short*)Ap;
    auto stage = [&](int kt) {
      const int b  = kt % 3;
      const int k0 = kt << 5;
      #pragma unroll
      for (int i = 0; i < 2; ++i) {
        int ch = tid + i * 128;
        int row = ch >> 2, koff = (ch & 3) * 8;
        int gm = mtile + row;
        int grow = (gm < M) ? gm : (M - 1);    // clamp: keep ASYNCcnt uniform
        const void* gp = (const void*)(A + (long)grow * lda + (k0 + koff));
        unsigned lp = lds_addr(&As[b][row * ALD + koff]);
        asm volatile("global_load_async_to_lds_b128 %0, %1, off"
                     :: "v"(lp), "v"(gp) : "memory");
      }
      #pragma unroll
      for (int i = 0; i < 2; ++i) {
        int ch = tid + i * 128;
        int row = ch >> 2, koff = (ch & 3) * 8;
        const void* gp = (const void*)(Bt + (long)(ntile + row) * ldb + (k0 + koff));
        unsigned lp = lds_addr(&Bs[b][row * BLD + koff]);
        asm volatile("global_load_async_to_lds_b128 %0, %1, off"
                     :: "v"(lp), "v"(gp) : "memory");
      }
    };
    stage(0);
    if (1 < nk) stage(1);
    for (int kt = 0; kt < nk; ++kt) {
      const int cur = kt % 3;
      // async ops complete in order: <=2 newer tiles outstanding => tile kt done
      if (kt + 2 < nk) {
        stage(kt + 2);
        asm volatile("s_wait_asynccnt 0x8" ::: "memory");
      } else if (kt + 1 < nk) {
        asm volatile("s_wait_asynccnt 0x4" ::: "memory");
      } else {
        asm volatile("s_wait_asynccnt 0x0" ::: "memory");
      }
      __syncthreads();
      compute(cur);
      __syncthreads();
    }
  } else {
    // ========== fp32-A: spill-free register-pipelined sync staging ==========
    // Two named register buffer sets; pipeline manually unrolled 2x so every
    // register-array index is a compile-time constant (no scratch demotion).
    const float* A = (const float*)Ap;
    float4 a0[4], a1[4];
    uint4  b0[2], b1[2];

    auto ld = [&](int kt, float4 (&fa)[4], uint4 (&fb)[2]) {
      const int k0 = kt << 5;
      if (k0 + 32 <= K) {              // uniform branch: no exec juggling
        #pragma unroll
        for (int i = 0; i < 4; ++i) {
          int ch = tid + i * 128;
          int row = ch >> 3, koff = (ch & 7) * 4;
          int gm = mtile + row;
          int grow = (gm < M) ? gm : (M - 1);
          fa[i] = *(const float4*)(A + (long)grow * lda + (k0 + koff));
        }
      } else {                         // K tail: chunks fully in or fully out
        #pragma unroll
        for (int i = 0; i < 4; ++i) {
          int ch = tid + i * 128;
          int row = ch >> 3, koff = (ch & 7) * 4;
          int gm = mtile + row, gk = k0 + koff;
          int grow = (gm < M) ? gm : (M - 1);
          float4 f = {0.f, 0.f, 0.f, 0.f};
          if (gk + 4 <= K) f = *(const float4*)(A + (long)grow * lda + gk);
          fa[i] = f;
        }
      }
      #pragma unroll
      for (int i = 0; i < 2; ++i) {
        int ch = tid + i * 128;
        int row = ch >> 2, koff = (ch & 3) * 8;
        fb[i] = *(const uint4*)(Bt + (long)(ntile + row) * ldb + (k0 + koff));
      }
    };
    auto stcvt = [&](int b, const float4 (&fa)[4], const uint4 (&fb)[2]) {
      #pragma unroll
      for (int i = 0; i < 4; ++i) {
        int ch = tid + i * 128;
        int row = ch >> 3, koff = (ch & 7) * 4;
        unsigned short r[4] = {f2bf(fa[i].x), f2bf(fa[i].y),
                               f2bf(fa[i].z), f2bf(fa[i].w)};
        *(uint2*)&As[b][row * ALD + koff] = *(const uint2*)r;
      }
      #pragma unroll
      for (int i = 0; i < 2; ++i) {
        int ch = tid + i * 128;
        int row = ch >> 2, koff = (ch & 3) * 8;
        *(uint4*)&Bs[b][row * BLD + koff] = fb[i];
      }
    };

    ld(0, a0, b0);
    for (int kt = 0; kt < nk; kt += 2) {
      if (kt + 1 < nk) ld(kt + 1, a1, b1);   // issue next tile's loads first
      stcvt(0, a0, b0);                      // partial loadcnt wait only
      __syncthreads();
      compute(0);
      __syncthreads();
      if (kt + 1 >= nk) break;
      if (kt + 2 < nk) ld(kt + 2, a0, b0);
      stcvt(1, a1, b1);
      __syncthreads();
      compute(1);
      __syncthreads();
    }
  }

  // ---- epilogue: lane owns column n, 8 contiguous rows per accumulator ----
  const int rsel = hi * 8;
  #pragma unroll
  for (int mi = 0; mi < 2; ++mi) {
    #pragma unroll
    for (int ni = 0; ni < 2; ++ni) {
      int mbase = mtile + wm + mi * 16 + rsel;   // rows mbase..mbase+7
      int n     = ntile + wn + ni * 16 + l15;
      v8f a = acc[mi][ni];
      if (bias) {
        float bv = bias[n];
        #pragma unroll
        for (int v = 0; v < 8; ++v) a[v] += bv;
      }
      if (relu) {
        #pragma unroll
        for (int v = 0; v < 8; ++v) a[v] = fmaxf(a[v], 0.f);
      }
      if (out_mode) {                            // Nc==64 for these calls
        float w = cw[depth * 64 + n];
        #pragma unroll
        for (int v = 0; v < 8; ++v) {
          int m = mbase + v;
          if (m < M) {
            long o = (long)m * 64 + n;
            if (out_mode == 1) out[o] = w * a[v];
            else               out[o] += w * a[v];
          }
        }
      }
      if (stT && mbase < M) {                    // M % 8 == 0 -> whole run valid
        BF8U b;
        #pragma unroll
        for (int v = 0; v < 8; ++v) b.u[v] = f2bf(a[v]);
        *(uint4*)&stT[(long)n * ldT + mbase] = b.q;
      }
      if (stRM) {
        #pragma unroll
        for (int v = 0; v < 8; ++v) {
          int m = mbase + v;
          if (m < M) stRM[(long)m * ldRM + n] = f2bf(a[v]);
        }
      }
    }
  }
}

// ---------------------------------------------------------------------------
// Helper kernels
// ---------------------------------------------------------------------------
__global__ void k_zero_u32(unsigned int* __restrict__ p, long n) {
  long i = (long)blockIdx.x * blockDim.x + threadIdx.x;
  if (i < n) p[i] = 0u;
}

// fp32 (M x K) -> bf16 (M x KP) with zero-padded K tail; 8 cols per thread
__global__ void k_convert_pad(const float* __restrict__ src,
                              unsigned short* __restrict__ dst,
                              int M, int K, int KP) {
  long idx = (long)blockIdx.x * blockDim.x + threadIdx.x;
  long per_row = KP >> 3;
  if (idx >= (long)M * per_row) return;
  int row = (int)(idx / per_row);
  int c8  = (int)(idx % per_row) * 8;
  unsigned short r[8];
  #pragma unroll
  for (int j = 0; j < 8; ++j) {
    int c = c8 + j;
    r[j] = f2bf((c < K) ? src[(long)row * K + c] : 0.f);
  }
  *(uint4*)&dst[(long)row * KP + c8] = *(const uint4*)r;
}

// fp32 src(R x C) row-major -> bf16 dst(C x R)  (dst leading dim = R)
__global__ void k_transpose_cvt(const float* __restrict__ src,
                                unsigned short* __restrict__ dst,
                                int R, int C) {
  long idx = (long)blockIdx.x * blockDim.x + threadIdx.x;
  if (idx >= (long)R * C) return;
  int r = (int)(idx / C), c = (int)(idx % C);
  dst[(long)c * R + r] = f2bf(src[idx]);
}

// ---------------------------------------------------------------------------
// Launch
// ---------------------------------------------------------------------------
extern "C" void kernel_launch(void* const* d_in, const int* in_sizes, int n_in,
                              void* d_out, int out_size, void* d_ws, size_t ws_size,
                              hipStream_t stream) {
  (void)in_sizes; (void)n_in; (void)out_size;
  const float* x   = (const float*)d_in[0];   // (N, 2048)
  const float* adj = (const float*)d_in[1];   // (N, N)
  const float* f0  = (const float*)d_in[2];   // (N, N)
  const float* f1  = (const float*)d_in[3];   // (N, N)
  const float* W1  = (const float*)d_in[4];   // (2048, 128)
  const float* b1  = (const float*)d_in[5];   // (128,)
  const float* W2  = (const float*)d_in[6];   // (128, 64)
  const float* b2  = (const float*)d_in[7];   // (64,)
  const float* cw  = (const float*)d_in[8];   // (1, 13, 64)
  float* out = (float*)d_out;                 // (N, 64) fp32

  const int N = 10000, NF = 2048, H = 128, C = 64;
  const int KP = 10016;                       // N rounded up to multiple of 32

  char* ws = (char*)d_ws;
  size_t off = 0;
  auto carve = [&](size_t bytes) -> char* {
    char* p = ws + off;
    off += (bytes + 255) & ~(size_t)255;
    return p;
  };
  unsigned short* hT  = (unsigned short*)carve((size_t)C * KP * 2); // h^T bf16
  unsigned short* pTa = (unsigned short*)carve((size_t)C * KP * 2); // p^T ping
  unsigned short* pTb = (unsigned short*)carve((size_t)C * KP * 2); // p^T pong
  unsigned short* h1  = (unsigned short*)carve((size_t)N * H * 2);  // relu(xW1+b1)
  unsigned short* W1t = (unsigned short*)carve((size_t)H * NF * 2); // W1^T bf16
  unsigned short* W2t = (unsigned short*)carve((size_t)C * H * 2);  // W2^T bf16
  unsigned short* nb  = (unsigned short*)(ws + off);                // bf16 adjacency
  const bool use_nb = (off + (size_t)N * KP * 2) <= ws_size;

  // 0) zero hT/pTa/pTb (contiguous) -> zero-padded K tails for Bt reads
  {
    long words = (3L * (size_t)C * KP * 2) / 4;
    k_zero_u32<<<dim3((unsigned)((words + 255) / 256)), dim3(256), 0, stream>>>(
        (unsigned int*)hT, words);
  }
  // 1) one-time bf16 materialization of the 10x-reused adjacency
  if (use_nb) {
    long thr = (long)N * (KP / 8);
    k_convert_pad<<<dim3((unsigned)((thr + 255) / 256)), dim3(256), 0, stream>>>(
        adj, nb, N, N, KP);
  }
  // 2) weight transposes (tiny)
  {
    long t1 = (long)NF * H;
    k_transpose_cvt<<<dim3((unsigned)((t1 + 255) / 256)), dim3(256), 0, stream>>>(
        W1, W1t, NF, H);
    long t2 = (long)H * C;
    k_transpose_cvt<<<dim3((unsigned)((t2 + 255) / 256)), dim3(256), 0, stream>>>(
        W2, W2t, H, C);
  }

  const dim3 blk(128);
  const dim3 gA((N + 63) / 64, H / 64);  // 157 x 2  (Nc = 128)
  const dim3 gB((N + 63) / 64, 1);       // 157 x 1  (Nc = 64)

  // 3) h1 = relu(x @ W1 + b1)    -> bf16 row-major
  gemm_wmma_bf16<true><<<gA, blk, 0, stream>>>(
      x, NF, W1t, NF, N, NF, b1, 1,
      nullptr, nullptr, 0, 0, nullptr, 0, h1, H);

  // 4) h = h1 @ W2 + b2 ; out = w[2]*h ; store h^T bf16
  gemm_wmma_bf16<false><<<gB, blk, 0, stream>>>(
      h1, H, W2t, H, N, H, b2, 0,
      out, cw, 2, 1, hT, KP, nullptr, 0);

  // 5) out += w[0]*(filt0 @ h), out += w[1]*(filt1 @ h)  (fp32 A, cvt-in-reg)
  gemm_wmma_bf16<true><<<gB, blk, 0, stream>>>(
      f0, N, hT, KP, N, N, nullptr, 0,
      out, cw, 0, 2, nullptr, 0, nullptr, 0);
  gemm_wmma_bf16<true><<<gB, blk, 0, stream>>>(
      f1, N, hT, KP, N, N, nullptr, 0,
      out, cw, 1, 2, nullptr, 0, nullptr, 0);

  // 6) ten power-iteration hops: p = A @ p ; out += w[2+k]*p
  unsigned short* src = hT;
  unsigned short* dsts[2] = {pTa, pTb};
  for (int k = 1; k <= 10; ++k) {
    unsigned short* dst = (k < 10) ? dsts[(k - 1) & 1] : nullptr;
    if (use_nb) {
      gemm_wmma_bf16<false><<<gB, blk, 0, stream>>>(
          nb, KP, src, KP, N, N, nullptr, 0,
          out, cw, 2 + k, 2, dst, KP, nullptr, 0);
    } else {
      gemm_wmma_bf16<true><<<gB, blk, 0, stream>>>(
          adj, N, src, KP, N, N, nullptr, 0,
          out, cw, 2 + k, 2, dst, KP, nullptr, 0);
    }
    src = dst;
  }
}
